// TransformerEncoderLayerWithRoPE_64922725646591
// MI455X (gfx1250) — matrix operand order repaired and checked by reading, not compiled, not run on hardware
//
#include <hip/hip_runtime.h>
#include <hip/hip_bf16.h>

#define DEV static __device__ __forceinline__

typedef __attribute__((ext_vector_type(16))) _Float16 v16h;
typedef __attribute__((ext_vector_type(8)))  _Float16 v8h;
typedef __attribute__((ext_vector_type(8)))  float    v8f;

constexpr int kB  = 16;
constexpr int kS  = 1024;
constexpr int kD  = 128;
constexpr int kH  = 8;
constexpr int kHD = 16;
constexpr int kF  = 512;
constexpr int kN  = kB * kS;            // 16384 tokens
constexpr float kEPS = 1.1920929e-07f;

DEV int lane_id() { return (int)(threadIdx.x & 31u); }

DEV v8f wmma16(v16h a, v16h b, v8f c) {
  // D = A(16x32 f16) * B(32x16 f16) + C(16x16 f32)
  return __builtin_amdgcn_wmma_f32_16x16x32_f16(false, a, false, b, (short)0, c,
                                                false, false);
}

// ---------------- WMMA operand tile loads ----------------
// A operand, full K=32, from row-major f16 [rows][...] with stride (halves).
DEV v16h load_a32(const _Float16* base, int stride) {
  const int l = lane_id();
  const _Float16* p = base + (size_t)(l & 15) * stride + ((l >> 4) << 3);
  v8h lo = *(const v8h*)p;          // K = koff .. koff+7
  v8h hi = *(const v8h*)(p + 16);   // K = 16+koff .. 16+koff+7
  v16h r;
#pragma unroll
  for (int i = 0; i < 8; ++i) { r[i] = lo[i]; r[i + 8] = hi[i]; }
  return r;
}

// A operand with real K=16, upper K half zero-padded.
DEV v16h load_a16_pad(const _Float16* base, int stride) {
  const int l = lane_id();
  const _Float16* p = base + (size_t)(l & 15) * stride + ((l >> 4) << 3);
  v8h lo = *(const v8h*)p;
  v16h r;
#pragma unroll
  for (int i = 0; i < 8; ++i) { r[i] = lo[i]; r[i + 8] = (_Float16)0.f; }
  return r;
}

// A operand, full K=32, from row-major f32 source (convert on the fly).
DEV v16h load_a32_f32(const float* base, int stride) {
  const int l = lane_id();
  const float* p = base + (size_t)(l & 15) * stride + ((l >> 4) << 3);
  v16h r;
#pragma unroll
  for (int i = 0; i < 8; ++i) {
    r[i]     = (_Float16)p[i];
    r[i + 8] = (_Float16)p[16 + i];
  }
  return r;
}

// B operand, full K=32, from B^T stored [n][k] row-major (stride in halves).
DEV v16h load_b32(const _Float16* baseT, int stride) {
  const int l = lane_id();
  const _Float16* p = baseT + (size_t)(l & 15) * stride + ((l >> 4) << 3);
  v8h lo = *(const v8h*)p;          // K = koff .. koff+7
  v8h hi = *(const v8h*)(p + 16);   // K = 16+koff .. 16+koff+7
  v16h r;
#pragma unroll
  for (int i = 0; i < 8; ++i) { r[i] = lo[i]; r[i + 8] = hi[i]; }
  return r;
}

// B operand with real K=16: elements 8..15 (K=16..31) are zero.
DEV v16h load_b16_pad(const _Float16* baseT, int stride) {
  const int l = lane_id();
  v16h r = {};
  {
    const _Float16* p = baseT + (size_t)(l & 15) * stride + ((l >> 4) << 3);
    v8h lo = *(const v8h*)p;
#pragma unroll
    for (int i = 0; i < 8; ++i) { r[i] = lo[i]; }
  }
  return r;
}

// ---------------- kernels ----------------

// Transpose+convert weight: w[rows][cols] f32 -> wT[cols][rows] f16
__global__ __launch_bounds__(256) void wcvt_kernel(const float* __restrict__ w,
                                                   _Float16* __restrict__ wT,
                                                   int rows, int cols) {
  int idx = blockIdx.x * 256 + threadIdx.x;
  if (idx < rows * cols) {
    int r = idx / cols, c = idx - r * cols;
    wT[(size_t)c * rows + r] = (_Float16)w[idx];
  }
}

// Fused QKV projection + bias + RoPE. One wave per (16-token tile, head).
// A tile reused across the q/k/v WMMAs each K-step.
// q16,k16: [N][D] f16 (head h occupies cols h*16..h*16+15)
// vT16:    [B][H][HD][S] f16 (transposed for direct B-operand use)
__global__ __launch_bounds__(32) void qkv_rope_kernel(
    const float* __restrict__ src,
    const _Float16* __restrict__ wqT, const _Float16* __restrict__ wkT,
    const _Float16* __restrict__ wvT,
    const float* __restrict__ bq, const float* __restrict__ bk,
    const float* __restrict__ bv,
    const float* __restrict__ cosT, const float* __restrict__ sinT,
    _Float16* __restrict__ q16, _Float16* __restrict__ k16,
    _Float16* __restrict__ vT16) {
  const int tile = blockIdx.x;          // N/16 tiles of 16 tokens
  const int h    = blockIdx.y;          // head
  const int l    = lane_id();
  const float* abase = src + (size_t)tile * 16 * kD;

  v8f cq = {}, ck = {}, cv = {};
#pragma unroll
  for (int k0 = 0; k0 < kD; k0 += 32) {
    v16h a = load_a32_f32(abase + k0, kD);
    cq = wmma16(a, load_b32(wqT + (size_t)(h * kHD) * kD + k0, kD), cq);
    ck = wmma16(a, load_b32(wkT + (size_t)(h * kHD) * kD + k0, kD), ck);
    cv = wmma16(a, load_b32(wvT + (size_t)(h * kHD) * kD + k0, kD), cv);
  }

  const int n = l & 15, hi = l >> 4;
  const float bqv = bq[h * kHD + n];
  const float bkv = bk[h * kHD + n];
  const float bvv = bv[h * kHD + n];
  const int d2 = n & 7;
#pragma unroll
  for (int j = 0; j < 8; ++j) {
    const int m = j + 8 * hi;
    const int t = tile * 16 + m;
    const int b = t >> 10;              // t / S
    const int s = t & (kS - 1);         // t % S
    const float cs = cosT[s * (kHD / 2) + d2];
    const float sn = sinT[s * (kHD / 2) + d2];
    float qv = cq[j] + bqv;
    float kv = ck[j] + bkv;
    float qp = __shfl_xor(qv, 8, 32);   // partner dim d^8 within the head
    float kp = __shfl_xor(kv, 8, 32);
    float qr = (n < 8) ? (qv * cs - qp * sn) : (qp * sn + qv * cs);
    float kr = (n < 8) ? (kv * cs - kp * sn) : (kp * sn + kv * cs);
    q16[(size_t)t * kD + h * kHD + n] = (_Float16)qr;
    k16[(size_t)t * kD + h * kHD + n] = (_Float16)kr;
    vT16[((size_t)(b * kH + h) * kHD + n) * kS + s] = (_Float16)(cv[j] + bvv);
  }
}

// Flash attention, one wave per (b, h, 16-row Q tile). ctx16: [N][D] f16.
__global__ __launch_bounds__(32) void attn_kernel(
    const _Float16* __restrict__ q16, const _Float16* __restrict__ k16,
    const _Float16* __restrict__ vT16, _Float16* __restrict__ ctx16) {
  const int qt = blockIdx.x;            // 0..S/16-1
  const int h  = blockIdx.y;            // 0..H-1
  const int b  = blockIdx.z;            // 0..B-1
  const int l  = lane_id();
  __shared__ _Float16 ldsP[16 * 16];

  const _Float16* qbase = q16 + ((size_t)(b * kS) + qt * 16) * kD + h * kHD;
  const v16h a = load_a16_pad(qbase, kD);
  const _Float16* vbase = vT16 + (size_t)(b * kH + h) * kHD * kS;

  float mrun[8], lrun[8];
  v8f acc = {};
#pragma unroll
  for (int j = 0; j < 8; ++j) { mrun[j] = -3.0e38f; lrun[j] = 0.f; }

  const int n = l & 15, hi = l >> 4;
  for (int kt = 0; kt < kS / 16; ++kt) {
    const _Float16* kbase = k16 + ((size_t)(b * kS) + kt * 16) * kD + h * kHD;
    v16h bk = load_b16_pad(kbase, kD);
    v8f sc = {};
    sc = wmma16(a, bk, sc);
#pragma unroll
    for (int j = 0; j < 8; ++j) {
      float x = sc[j] * 0.25f;          // 1/sqrt(HD)
      float mx = x;
      mx = fmaxf(mx, __shfl_xor(mx, 1, 32));
      mx = fmaxf(mx, __shfl_xor(mx, 2, 32));
      mx = fmaxf(mx, __shfl_xor(mx, 4, 32));
      mx = fmaxf(mx, __shfl_xor(mx, 8, 32));
      float mnew = fmaxf(mrun[j], mx);
      float p    = __expf(x - mnew);
      float corr = __expf(mrun[j] - mnew);
      float ps = p;
      ps += __shfl_xor(ps, 1, 32);
      ps += __shfl_xor(ps, 2, 32);
      ps += __shfl_xor(ps, 4, 32);
      ps += __shfl_xor(ps, 8, 32);
      lrun[j] = lrun[j] * corr + ps;
      mrun[j] = mnew;
      acc[j] *= corr;
      ldsP[(j + 8 * hi) * 16 + n] = (_Float16)p;   // C-layout -> [m][n]
    }
    __syncthreads();
    v16h pa = load_a16_pad(ldsP, 16);              // reload P as A operand
    v16h vb = load_b16_pad(vbase + kt * 16, kS);   // V^T rows = head dims
    acc = wmma16(pa, vb, acc);
    __syncthreads();
  }
#pragma unroll
  for (int j = 0; j < 8; ++j) {
    const int t = b * kS + qt * 16 + j + 8 * hi;
    float r = acc[j] * __frcp_rn(lrun[j]);
    ctx16[(size_t)t * kD + h * kHD + n] = (_Float16)r;
  }
}

// Output projection + bias + residual -> xpre (f32 [N][D])
// One wave -> 16x64 output (4 column tiles, A reused).
__global__ __launch_bounds__(32) void oproj_kernel(
    const _Float16* __restrict__ ctx16, const _Float16* __restrict__ woT,
    const float* __restrict__ bo, const float* __restrict__ src,
    float* __restrict__ xpre) {
  const int tile = blockIdx.x, jb = blockIdx.y;   // jb in 0..D/64-1
  v8f c[4] = {};
#pragma unroll
  for (int k0 = 0; k0 < kD; k0 += 32) {
    v16h a = load_a32(ctx16 + (size_t)tile * 16 * kD + k0, kD);
#pragma unroll
    for (int u = 0; u < 4; ++u)
      c[u] = wmma16(a, load_b32(woT + (size_t)((jb * 4 + u) * 16) * kD + k0, kD),
                    c[u]);
  }
  const int l = lane_id(), n = l & 15, hi = l >> 4;
#pragma unroll
  for (int u = 0; u < 4; ++u) {
    const int col = (jb * 4 + u) * 16 + n;
    const float bv = bo[col];
#pragma unroll
    for (int j = 0; j < 8; ++j) {
      const int t = tile * 16 + j + 8 * hi;
      const size_t idx = (size_t)t * kD + col;
      xpre[idx] = c[u][j] + bv + src[idx];
    }
  }
}

// RMSNorm over D=128; one block (128 threads) per token.
__global__ __launch_bounds__(128) void rmsnorm_kernel(
    const float* __restrict__ xin, const float* __restrict__ g,
    float* __restrict__ out32, _Float16* __restrict__ out16) {
  const int t = blockIdx.x;
  const int i = threadIdx.x;
  const float v = xin[(size_t)t * kD + i];
  float ss = v * v;
  ss += __shfl_xor(ss, 1, 32);
  ss += __shfl_xor(ss, 2, 32);
  ss += __shfl_xor(ss, 4, 32);
  ss += __shfl_xor(ss, 8, 32);
  ss += __shfl_xor(ss, 16, 32);
  __shared__ float red[4];
  if ((i & 31) == 0) red[i >> 5] = ss;
  __syncthreads();
  const float tot = red[0] + red[1] + red[2] + red[3];
  const float scale = rsqrtf(tot * (1.0f / kD) + kEPS);
  const float r = v * scale * g[i];
  if (out32) out32[(size_t)t * kD + i] = r;
  if (out16) out16[(size_t)t * kD + i] = (_Float16)r;
}

// FFN up-projection + bias + exact GELU -> h16 (f16 [N][F])
// One wave -> 16x64 output (4 column tiles, A reused).
__global__ __launch_bounds__(32) void ffn1_kernel(
    const _Float16* __restrict__ x16, const _Float16* __restrict__ w1T,
    const float* __restrict__ b1, _Float16* __restrict__ h16) {
  const int tile = blockIdx.x, jb = blockIdx.y;   // jb in 0..F/64-1
  v8f c[4] = {};
#pragma unroll
  for (int k0 = 0; k0 < kD; k0 += 32) {
    v16h a = load_a32(x16 + (size_t)tile * 16 * kD + k0, kD);
#pragma unroll
    for (int u = 0; u < 4; ++u)
      c[u] = wmma16(a, load_b32(w1T + (size_t)((jb * 4 + u) * 16) * kD + k0, kD),
                    c[u]);
  }
  const int l = lane_id(), n = l & 15, hi = l >> 4;
#pragma unroll
  for (int u = 0; u < 4; ++u) {
    const int col = (jb * 4 + u) * 16 + n;
    const float bv = b1[col];
#pragma unroll
    for (int j = 0; j < 8; ++j) {
      const int t = tile * 16 + j + 8 * hi;
      const float val = c[u][j] + bv;
      const float ge = 0.5f * val * (1.0f + erff(val * 0.70710678118654752f));
      h16[(size_t)t * kF + col] = (_Float16)ge;
    }
  }
}

// FFN down-projection + bias + residual(x) -> ffpre (f32 [N][D])
// One wave -> 16x64 output (4 column tiles, A reused). K = 512.
__global__ __launch_bounds__(32) void ffn2_kernel(
    const _Float16* __restrict__ h16, const _Float16* __restrict__ w2T,
    const float* __restrict__ b2, const float* __restrict__ x32,
    float* __restrict__ ffpre) {
  const int tile = blockIdx.x, jb = blockIdx.y;   // jb in 0..D/64-1
  v8f c[4] = {};
#pragma unroll 4
  for (int k0 = 0; k0 < kF; k0 += 32) {
    v16h a = load_a32(h16 + (size_t)tile * 16 * kF + k0, kF);
#pragma unroll
    for (int u = 0; u < 4; ++u)
      c[u] = wmma16(a, load_b32(w2T + (size_t)((jb * 4 + u) * 16) * kF + k0, kF),
                    c[u]);
  }
  const int l = lane_id(), n = l & 15, hi = l >> 4;
#pragma unroll
  for (int u = 0; u < 4; ++u) {
    const int col = (jb * 4 + u) * 16 + n;
    const float bv = b2[col];
#pragma unroll
    for (int j = 0; j < 8; ++j) {
      const int t = tile * 16 + j + 8 * hi;
      const size_t idx = (size_t)t * kD + col;
      ffpre[idx] = c[u][j] + bv + x32[idx];
    }
  }
}

// ---------------- workspace layout (bytes) ----------------
constexpr size_t OFF_WQT  = 0;
constexpr size_t OFF_WKT  = OFF_WQT + (size_t)kD * kD * 2;      //  32768
constexpr size_t OFF_WVT  = OFF_WKT + (size_t)kD * kD * 2;      //  65536
constexpr size_t OFF_WOT  = OFF_WVT + (size_t)kD * kD * 2;      //  98304
constexpr size_t OFF_W1T  = OFF_WOT + (size_t)kD * kD * 2;      // 131072
constexpr size_t OFF_W2T  = OFF_W1T + (size_t)kD * kF * 2;      // 262144
constexpr size_t OFF_Q16  = OFF_W2T + (size_t)kF * kD * 2;      // 393216
constexpr size_t OFF_K16  = OFF_Q16 + (size_t)kN * kD * 2;
constexpr size_t OFF_VT   = OFF_K16 + (size_t)kN * kD * 2;
constexpr size_t OFF_CTX  = OFF_VT  + (size_t)kN * kD * 2;
constexpr size_t OFF_XPRE = OFF_CTX + (size_t)kN * kD * 2;
constexpr size_t OFF_X32  = OFF_XPRE + (size_t)kN * kD * 4;
constexpr size_t OFF_X16  = OFF_X32 + (size_t)kN * kD * 4;
// aliases (safe via stream ordering): h16 reuses q/k/v/ctx; ffpre reuses xpre
constexpr size_t OFF_H16   = OFF_Q16;    // needs N*F*2 = 16 MB == q..ctx region
constexpr size_t OFF_FFPRE = OFF_XPRE;

extern "C" void kernel_launch(void* const* d_in, const int* in_sizes, int n_in,
                              void* d_out, int out_size, void* d_ws, size_t ws_size,
                              hipStream_t stream) {
  (void)in_sizes; (void)n_in; (void)out_size; (void)ws_size;
  const float* src  = (const float*)d_in[0];
  const float* cosT = (const float*)d_in[1];
  const float* sinT = (const float*)d_in[2];
  const float* wq = (const float*)d_in[3];
  const float* bq = (const float*)d_in[4];
  const float* wk = (const float*)d_in[5];
  const float* bk = (const float*)d_in[6];
  const float* wv = (const float*)d_in[7];
  const float* bv = (const float*)d_in[8];
  const float* wo = (const float*)d_in[9];
  const float* bo = (const float*)d_in[10];
  const float* w1 = (const float*)d_in[11];
  const float* b1 = (const float*)d_in[12];
  const float* w2 = (const float*)d_in[13];
  const float* b2 = (const float*)d_in[14];
  const float* g1 = (const float*)d_in[15];
  const float* g2 = (const float*)d_in[16];
  float* out = (float*)d_out;

  char* ws = (char*)d_ws;
  _Float16* wqT  = (_Float16*)(ws + OFF_WQT);
  _Float16* wkT  = (_Float16*)(ws + OFF_WKT);
  _Float16* wvT  = (_Float16*)(ws + OFF_WVT);
  _Float16* woT  = (_Float16*)(ws + OFF_WOT);
  _Float16* w1T  = (_Float16*)(ws + OFF_W1T);
  _Float16* w2T  = (_Float16*)(ws + OFF_W2T);
  _Float16* q16  = (_Float16*)(ws + OFF_Q16);
  _Float16* k16  = (_Float16*)(ws + OFF_K16);
  _Float16* vT16 = (_Float16*)(ws + OFF_VT);
  _Float16* ctx16 = (_Float16*)(ws + OFF_CTX);
  float*    xpre  = (float*)(ws + OFF_XPRE);
  float*    x32   = (float*)(ws + OFF_X32);
  _Float16* x16   = (_Float16*)(ws + OFF_X16);
  _Float16* h16   = (_Float16*)(ws + OFF_H16);
  float*    ffpre = (float*)(ws + OFF_FFPRE);

  // 1) weights -> transposed f16
  wcvt_kernel<<<(kD * kD + 255) / 256, 256, 0, stream>>>(wq, wqT, kD, kD);
  wcvt_kernel<<<(kD * kD + 255) / 256, 256, 0, stream>>>(wk, wkT, kD, kD);
  wcvt_kernel<<<(kD * kD + 255) / 256, 256, 0, stream>>>(wv, wvT, kD, kD);
  wcvt_kernel<<<(kD * kD + 255) / 256, 256, 0, stream>>>(wo, woT, kD, kD);
  wcvt_kernel<<<(kD * kF + 255) / 256, 256, 0, stream>>>(w1, w1T, kD, kF);
  wcvt_kernel<<<(kF * kD + 255) / 256, 256, 0, stream>>>(w2, w2T, kF, kD);

  // 2) QKV + RoPE
  qkv_rope_kernel<<<dim3(kN / 16, kH), 32, 0, stream>>>(
      src, wqT, wkT, wvT, bq, bk, bv, cosT, sinT, q16, k16, vT16);

  // 3) flash attention
  attn_kernel<<<dim3(kS / 16, kH, kB), 32, 0, stream>>>(q16, k16, vT16, ctx16);

  // 4) output projection + residual (16x64 per wave)
  oproj_kernel<<<dim3(kN / 16, kD / 64), 32, 0, stream>>>(ctx16, woT, bo, src, xpre);

  // 5) RMSNorm 1 -> x (f32) and x16 (f16)
  rmsnorm_kernel<<<kN, 128, 0, stream>>>(xpre, g1, x32, x16);

  // 6) FFN up + GELU (16x64 per wave)
  ffn1_kernel<<<dim3(kN / 16, kF / 64), 32, 0, stream>>>(x16, w1T, b1, h16);

  // 7) FFN down + residual (16x64 per wave)
  ffn2_kernel<<<dim3(kN / 16, kD / 64), 32, 0, stream>>>(h16, w2T, b2, x32, ffpre);

  // 8) RMSNorm 2 -> final output (f32)
  rmsnorm_kernel<<<kN, 128, 0, stream>>>(ffpre, g2, out, nullptr);
}
